// _SymbolicPredicateGrounding_75058848465012
// MI455X (gfx1250) — matrix operand order repaired and checked by reading, not compiled, not run on hardware
//
#include <hip/hip_runtime.h>

#define B_   4096
#define F_   1024
#define P_   64
#define H1_  256
#define H2_  128

typedef __attribute__((ext_vector_type(16))) __bf16        v16bf;
typedef __attribute__((ext_vector_type(8)))  float         v8f;
typedef __attribute__((ext_vector_type(4)))  float         f4;
typedef __attribute__((ext_vector_type(4)))  unsigned int  u4;

union FragU { u4 q[2]; unsigned int u[8]; v16bf v; };

// round-half-up bf16 pack: 3 VALU (add, add, v_perm_b32)
__device__ __forceinline__ unsigned int pack_bf16x2(float lo, float hi) {
    unsigned int ul = __float_as_uint(lo) + 0x8000u;
    unsigned int uh = __float_as_uint(hi) + 0x8000u;
    return __builtin_amdgcn_perm(uh, ul, 0x07060302u);  // {uh[31:16], ul[31:16]}
}
__device__ __forceinline__ unsigned short f32_to_bf16(float f) {
    return (unsigned short)((__float_as_uint(f) + 0x8000u) >> 16);
}
__device__ __forceinline__ float bf16_to_f32(unsigned short s) {
    return __uint_as_float(((unsigned int)s) << 16);
}

#define LDA1 40   // bf16 stride (80 B, 16B-aligned, 20-bank step: conflict-free b128)
#define LDB  40   // bf16 stride for staged B tiles [N][32]
#define LDH1 264  // bf16 stride for h1 (528 B, 16B-aligned)
#define LDH2 132  // bf16 stride for h2

// A-fragment (16-bit 16x32, ISA 7.12.2): lane-half 0 rows hold K{0..7,16..23},
// half 1 holds K{8..15,24..31}; two 16B loads.
__device__ __forceinline__ v16bf load_fragA(const unsigned short* t, int stride,
                                            int rbase, int lane, int kofs) {
    FragU f;
    const unsigned short* rp = t + (rbase + (lane & 15)) * stride + kofs + ((lane & 16) >> 1);
    f.q[0] = *(const u4*)rp;          // K pairs 0..7
    f.q[1] = *(const u4*)(rp + 16);   // K pairs 16..23
    return f.v;
}

// B-fragment (32x16, column per lane): half 0 holds K{0..7,16..23}, half 1 K{8..15,24..31}.
__device__ __forceinline__ v16bf load_fragB(const unsigned short* t, int stride,
                                            int cbase, int lane) {
    FragU f;
    const unsigned short* rp = t + (cbase + (lane & 15)) * stride + ((lane & 16) >> 1);
    f.q[0] = *(const u4*)rp;          // K pairs 0..7
    f.q[1] = *(const u4*)(rp + 16);   // K pairs 16..23
    return f.v;
}

__device__ __forceinline__ v8f wmma_bf16(v16bf a, v16bf b, v8f c) {
    return __builtin_amdgcn_wmma_f32_16x16x32_bf16(false, a, false, b,
                                                   (short)0, c, false, false);
}

__global__ __launch_bounds__(256) void
fused_pred_mlp(const float* __restrict__ feat, const float* __restrict__ W1,
               const float* __restrict__ b1,   const float* __restrict__ W2,
               const float* __restrict__ b2,   const float* __restrict__ W3,
               const float* __restrict__ b3,   float* __restrict__ out) {
    __shared__ __align__(16) unsigned char smem[59392];
    unsigned short* lds_a  = (unsigned short*)smem;            // 64*40*2   = 5120 B
    unsigned short* lds_b  = (unsigned short*)(smem + 5120);   // 256*40*2  = 20480 B
    unsigned short* lds_h2 = (unsigned short*)smem;            // 64*132*2  = 16896 B (alias)
    unsigned short* lds_h1 = (unsigned short*)(smem + 25600);  // 64*264*2  = 33792 B

    const int tid  = threadIdx.x;
    const int lane = tid & 31;
    const int wave = tid >> 5;
    const int p        = blockIdx.x >> 6;        // predicate
    const int blockRow = (blockIdx.x & 63) << 6; // 64-row B tile

    // ---------------- Layer 1: [64,1024] x [1024,256] ----------------
    const int wm = wave >> 2;   // 0..1  -> 32 rows
    const int wn = wave & 3;    // 0..3  -> 64 cols
    v8f acc1[2][4];
#pragma unroll
    for (int mt = 0; mt < 2; ++mt)
#pragma unroll
        for (int nt = 0; nt < 4; ++nt)
            acc1[mt][nt] = (v8f){0.f,0.f,0.f,0.f,0.f,0.f,0.f,0.f};

    for (int kc = 0; kc < F_ / 32; ++kc) {
        __syncthreads();
        {   // stage A: features 64x32 -> bf16 LDS, one b128 store per thread
            int r  = tid >> 2;
            int cg = (tid & 3) << 3;
            const float* g = feat + (size_t)(blockRow + r) * F_ + kc * 32 + cg;
            f4 x0 = *(const f4*)g;
            f4 x1 = *(const f4*)(g + 4);
            u4 pk;
            pk[0] = pack_bf16x2(x0[0], x0[1]);
            pk[1] = pack_bf16x2(x0[2], x0[3]);
            pk[2] = pack_bf16x2(x1[0], x1[1]);
            pk[3] = pack_bf16x2(x1[2], x1[3]);
            *(u4*)(lds_a + r * LDA1 + cg) = pk;
        }
        {   // stage B: W1[p] 32x256 K-gather transpose; thread t owns column n=t.
            // Lane-consecutive n -> each of the 32 b32 loads is wave-coalesced.
            const float* gw = W1 + (size_t)(p * F_ + kc * 32) * H1_ + tid;
            unsigned short* db = lds_b + tid * LDB;
#pragma unroll
            for (int kk = 0; kk < 32; kk += 8) {
                u4 pk;
#pragma unroll
                for (int j = 0; j < 4; ++j)
                    pk[j] = pack_bf16x2(gw[(size_t)(kk + 2 * j) * H1_],
                                        gw[(size_t)(kk + 2 * j + 1) * H1_]);
                *(u4*)(db + kk) = pk;
            }
        }
        __syncthreads();

        v16bf afr[2], bfr[4];
#pragma unroll
        for (int mt = 0; mt < 2; ++mt)
            afr[mt] = load_fragA(lds_a, LDA1, wm * 32 + mt * 16, lane, 0);
#pragma unroll
        for (int nt = 0; nt < 4; ++nt)
            bfr[nt] = load_fragB(lds_b, LDB, wn * 64 + nt * 16, lane);
#pragma unroll
        for (int mt = 0; mt < 2; ++mt)
#pragma unroll
            for (int nt = 0; nt < 4; ++nt)
                acc1[mt][nt] = wmma_bf16(afr[mt], bfr[nt], acc1[mt][nt]);
    }

    // epilogue 1: +b1, relu, bf16 -> lds_h1
#pragma unroll
    for (int mt = 0; mt < 2; ++mt)
#pragma unroll
        for (int nt = 0; nt < 4; ++nt) {
            int col = wn * 64 + nt * 16 + (lane & 15);
            float bias = b1[p * H1_ + col];
            int mrow = wm * 32 + mt * 16 + ((lane & 16) >> 1);
#pragma unroll
            for (int i = 0; i < 8; ++i) {
                float v = acc1[mt][nt][i] + bias;
                v = v > 0.f ? v : 0.f;
                lds_h1[(mrow + i) * LDH1 + col] = f32_to_bf16(v);
            }
        }

    // ---------------- Layer 2: [64,256] x [256,128] ----------------
    const int wm2 = wave >> 1;  // 0..3 -> 16 rows
    const int wn2 = wave & 1;   // 0..1 -> 64 cols
    v8f acc2[4];
#pragma unroll
    for (int nt = 0; nt < 4; ++nt)
        acc2[nt] = (v8f){0.f,0.f,0.f,0.f,0.f,0.f,0.f,0.f};

    for (int kc = 0; kc < H1_ / 32; ++kc) {
        __syncthreads();   // also publishes lds_h1 on first iteration
        {   // stage W2[p] 32x128 K-gather transpose: 2 threads per column
            int n  = tid & 127;
            int kh = (tid >> 7) << 4;   // 0 or 16
            const float* gw = W2 + (size_t)(p * H1_ + kc * 32 + kh) * H2_ + n;
            unsigned short* db = lds_b + n * LDB + kh;
#pragma unroll
            for (int kk = 0; kk < 16; kk += 8) {
                u4 pk;
#pragma unroll
                for (int j = 0; j < 4; ++j)
                    pk[j] = pack_bf16x2(gw[(size_t)(kk + 2 * j) * H2_],
                                        gw[(size_t)(kk + 2 * j + 1) * H2_]);
                *(u4*)(db + kk) = pk;
            }
        }
        __syncthreads();

        v16bf af = load_fragA(lds_h1, LDH1, wm2 * 16, lane, kc * 32);
#pragma unroll
        for (int nt = 0; nt < 4; ++nt) {
            v16bf bf = load_fragB(lds_b, LDB, wn2 * 64 + nt * 16, lane);
            acc2[nt] = wmma_bf16(af, bf, acc2[nt]);
        }
    }

    __syncthreads();  // lds_h2 aliases lds_a/lds_b: wait for all B-frag reads
    // epilogue 2: +b2, relu, bf16 -> lds_h2
#pragma unroll
    for (int nt = 0; nt < 4; ++nt) {
        int col = wn2 * 64 + nt * 16 + (lane & 15);
        float bias = b2[p * H2_ + col];
        int mrow = wm2 * 16 + ((lane & 16) >> 1);
#pragma unroll
        for (int i = 0; i < 8; ++i) {
            float v = acc2[nt][i] + bias;
            v = v > 0.f ? v : 0.f;
            lds_h2[(mrow + i) * LDH2 + col] = f32_to_bf16(v);
        }
    }
    __syncthreads();

    // ---------------- Layer 3: [64,128] . [128] + sigmoid ----------------
    {
        int row  = tid >> 2;
        int part = tid & 3;
        const unsigned short* h2r = lds_h2 + row * LDH2 + part * 32;
        const float* w3 = W3 + p * H2_ + part * 32;
        float s = 0.f;
#pragma unroll
        for (int j = 0; j < 32; ++j)
            s += bf16_to_f32(h2r[j]) * w3[j];
        s += __shfl_xor(s, 1, 32);
        s += __shfl_xor(s, 2, 32);
        if (part == 0) {
            float logit = s + b3[p];
            out[(size_t)(blockRow + row) * P_ + p] = 1.f / (1.f + __expf(-logit));
        }
    }
}

extern "C" void kernel_launch(void* const* d_in, const int* in_sizes, int n_in,
                              void* d_out, int out_size, void* d_ws, size_t ws_size,
                              hipStream_t stream) {
    (void)in_sizes; (void)n_in; (void)out_size; (void)d_ws; (void)ws_size;
    const float* feat = (const float*)d_in[0];
    const float* W1   = (const float*)d_in[1];
    const float* b1   = (const float*)d_in[2];
    const float* W2   = (const float*)d_in[3];
    const float* b2   = (const float*)d_in[4];
    const float* W3   = (const float*)d_in[5];
    const float* b3   = (const float*)d_in[6];
    float* out = (float*)d_out;

    dim3 grid(P_ * (B_ / 64));   // 4096 blocks: predicate x 64-row tile
    dim3 block(256);             // 8 wave32s
    fused_pred_mlp<<<grid, block, 0, stream>>>(feat, W1, b1, W2, b2, W3, b3, out);
}
